// TinyTransformerBlock_71150428225553
// MI455X (gfx1250) — matrix operand order repaired and checked by reading, not compiled, not run on hardware
//
#include <hip/hip_runtime.h>
#include <hip/hip_bf16.h>
#include <math.h>

typedef __bf16 bf16_t;
typedef __attribute__((ext_vector_type(16))) __bf16 v16bf;
typedef __attribute__((ext_vector_type(8)))  __bf16 v8bf;
typedef __attribute__((ext_vector_type(8)))  float   v8f;
typedef __attribute__((ext_vector_type(4)))  unsigned int v4u;
typedef __attribute__((ext_vector_type(8)))  int v8i_t;
typedef __attribute__((ext_vector_type(4)))  int v4i_t;

// ---------------------------------------------------------------------------
// WMMA helpers (CDNA5 16x16x32 bf16, f32 accumulate)
// ---------------------------------------------------------------------------
static __device__ inline v8f wmma_bf16(v16bf a, v16bf b, v8f c) {
  return __builtin_amdgcn_wmma_f32_16x16x32_bf16(false, a, false, b, (short)0, c,
                                                 false, false);
}

// A-fragment (16 rows x 32 K), source row-major with leading dim ld.
// Lane L (0..15): row=L, K in {kb..kb+7, kb+16..kb+23}, kb = (L>=16)?8:0.
static __device__ inline v16bf load_frag_a(const bf16_t* p, int ld) {
  int lane = threadIdx.x & 31;
  const bf16_t* r = p + (size_t)(lane & 15) * ld + ((lane >> 4) << 3);
  v16bf out;
  ((v8bf*)&out)[0] = *(const v8bf*)(r);
  ((v8bf*)&out)[1] = *(const v8bf*)(r + 16);
  return out;
}

// B-fragment (32 K x 16 cols) read from K-contiguous ("B^T") storage p[n*ld+k].
// Lane L: col=L&15, K in {kb..kb+7, kb+16..kb+23}, kb = (L>=16)?8:0.
static __device__ inline v16bf load_frag_bt(const bf16_t* p, int ld) {
  int lane = threadIdx.x & 31;
  const bf16_t* r = p + (size_t)(lane & 15) * ld + ((lane >> 4) << 3);
  v16bf out;
  ((v8bf*)&out)[0] = *(const v8bf*)(r);
  ((v8bf*)&out)[1] = *(const v8bf*)(r + 16);
  return out;
}

// ---------------------------------------------------------------------------
// Tensor Data Mover: DMA a 128-row x 32-col bf16 tile (row stride K elements)
// from global memory into LDS.  D# layout per CDNA5 ISA ch.8:
//   group0: {count=1, lds_addr, global_addr[56:0], type=2}
//   group1: {data_size=1(2B), tensor_dim0/1, tile_dim0=32, tile_dim1=128,
//            tensor_dim0_stride=K}
//   groups 2/3: zero (2-D tensor)
// ---------------------------------------------------------------------------
#if __has_builtin(__builtin_amdgcn_tensor_load_to_lds)
#define HAVE_TDM 1
static __device__ inline void tdm_load_tile_a(const bf16_t* gsrc, unsigned lds_off,
                                              int rem_k, int rem_m, int strideK) {
  unsigned long long ga = (unsigned long long)(uintptr_t)gsrc;
  v4u g0 = {1u,                                   // count=1, user descriptor
            lds_off,                              // LDS byte address
            (unsigned)(ga & 0xffffffffu),         // global_addr[31:0]
            (unsigned)((ga >> 32) & 0x01ffffffu) | (2u << 30)};  // [56:32]|type=2
  int w1 = (rem_k & 0xffff) << 16;                              // dim0 lo16
  int w2 = ((rem_k >> 16) & 0xffff) | ((rem_m & 0xffff) << 16); // dim0 hi/dim1 lo
  int w3 = ((rem_m >> 16) & 0xffff) | (32 << 16);               // dim1 hi|tile0=32
  v8i_t g1 = {0x00010000,  // data_size=1 (2 bytes); no multicast/pad/iterate
              w1, w2, w3,
              128,         // tile_dim1=128 rows, tile_dim2=0
              strideK,     // tensor_dim0_stride lo32
              0, 0};
  v4i_t gz = {0, 0, 0, 0};
#if defined(__clang_major__) && (__clang_major__ >= 23)
  v8i_t gz8 = {0, 0, 0, 0, 0, 0, 0, 0};
  __builtin_amdgcn_tensor_load_to_lds(g0, g1, gz, gz, gz8, 0);
#else
  __builtin_amdgcn_tensor_load_to_lds(g0, g1, gz, gz, 0);
#endif
}
#else
#define HAVE_TDM 0
#endif

// ---------------------------------------------------------------------------
// Tiled GEMM: C[M,N] = A[M,K](bf16,rm) * W[K,N](bf16,rm) (+bias)(+resid)(GELU)
// Block: 256 threads (8 waves), tile 128x64, waves in 4x2 grid of 32x32.
// A tile moved global->LDS by the Tensor Data Mover (wave 0 issues, waits
// TENSORcnt, workgroup barrier publishes). W tile staged transposed by VALU.
// Requires M%128==0, K%32==0; N guarded (for LoRA N=16).
// ---------------------------------------------------------------------------
template <bool GELU>
__global__ __launch_bounds__(256)
void gemm_bf16_kernel(const bf16_t* __restrict__ A, const bf16_t* __restrict__ W,
                      const float* __restrict__ bias, const float* __restrict__ resid,
                      float* __restrict__ outF, bf16_t* __restrict__ outB,
                      int M, int N, int K, int ldc) {
  __shared__ alignas(32) bf16_t sA[128 * 32];
  __shared__ alignas(32) bf16_t sBt[64 * 32];  // transposed: sBt[n][k]

  const int n0 = blockIdx.x * 64;
  const int m0 = blockIdx.y * 128;
  const int tid = threadIdx.x;
  const int wave = tid >> 5;
  const int lane = tid & 31;
  const int wr = wave >> 1;  // 0..3
  const int wc = wave & 1;   // 0..1

  v8f acc[2][2] = {};

  const int wk = tid >> 3;            // 0..31
  const int wn = (tid & 7) << 3;      // 0..56
#if !HAVE_TDM
  const int arow = tid >> 1;
  const int ahalf = (tid & 1) << 4;
#endif
  const unsigned sA_lds = (unsigned)(uintptr_t)(void*)sA;  // LDS aperture lo32

  for (int k0 = 0; k0 < K; k0 += 32) {
#if HAVE_TDM
    // wave 0: async DMA of the 128x32 A tile straight into LDS
    if (tid < 32)
      tdm_load_tile_a(A + (size_t)m0 * K + k0, sA_lds, K - k0, M - m0, K);
#else
    *(v16bf*)(sA + arow * 32 + ahalf) =
        *(const v16bf*)(A + (size_t)(m0 + arow) * K + k0 + ahalf);
#endif
    // stage W tile 32x64, transposed into sBt[n][k]
    v8bf wv = {};
    if (n0 + wn < N)
      wv = *(const v8bf*)(W + (size_t)(k0 + wk) * N + n0 + wn);
#pragma unroll
    for (int j = 0; j < 8; ++j) sBt[(wn + j) * 32 + wk] = wv[j];
    if (k0 + 32 < K)  // hint next W tile into cache (global_prefetch_b8)
      __builtin_prefetch(W + (size_t)(k0 + 32 + wk) * N + n0 + wn, 0, 0);
#if HAVE_TDM
    if (tid < 32) __builtin_amdgcn_s_wait_tensorcnt(0);
    // Compiler fence: the TDM wrote sA "behind the optimizer's back" (the LDS
    // address travels inside the descriptor, not as a pointer operand). The
    // opaque memory clobber makes sA's contents unanalyzable so the fragment
    // loads + WMMAs below cannot be undef-folded/DCE'd.
    __asm__ volatile("" ::: "memory");
#endif
    __syncthreads();

    const bf16_t* pa = sA + wr * 32 * 32;
    const bf16_t* pb = sBt + wc * 32 * 32;
    v16bf a0 = load_frag_a(pa, 32);
    v16bf a1 = load_frag_a(pa + 16 * 32, 32);
    v16bf b0 = load_frag_bt(pb, 32);
    v16bf b1 = load_frag_bt(pb + 16 * 32, 32);
    acc[0][0] = wmma_bf16(a0, b0, acc[0][0]);
    acc[0][1] = wmma_bf16(a0, b1, acc[0][1]);
    acc[1][0] = wmma_bf16(a1, b0, acc[1][0]);
    acc[1][1] = wmma_bf16(a1, b1, acc[1][1]);
    __syncthreads();
  }

  // Epilogue. C layout: lane holds col=lane&15, rows rb..rb+7 (rb = hi-half*8).
  const int rb = (lane >> 4) << 3;
  const int cl = lane & 15;
#pragma unroll
  for (int ti = 0; ti < 2; ++ti) {
#pragma unroll
    for (int tj = 0; tj < 2; ++tj) {
      int col = n0 + wc * 32 + tj * 16 + cl;
      if (col >= N) continue;
      float bb = bias ? bias[col] : 0.0f;
#pragma unroll
      for (int i = 0; i < 8; ++i) {
        int r = m0 + wr * 32 + ti * 16 + rb + i;
        float v = acc[ti][tj][i] + bb;
        if (resid) v += resid[(size_t)r * ldc + col];
        if (GELU) v = 0.5f * v * (1.0f + erff(v * 0.70710678118f));
        if (outF) outF[(size_t)r * ldc + col] = v;
        if (outB) outB[(size_t)r * ldc + col] = (bf16_t)v;
      }
    }
  }
}

// ---------------------------------------------------------------------------
// Flash attention, d_head=64, H=12, layouts [B, seq, H*64] bf16.
// Block = 128 threads (4 waves x 16 queries), key chunks of 32.
// ---------------------------------------------------------------------------
__global__ __launch_bounds__(128)
void flash_attn_kernel(const bf16_t* __restrict__ Q, const bf16_t* __restrict__ Km,
                       const bf16_t* __restrict__ Vm, const float* __restrict__ gates,
                       bf16_t* __restrict__ Z, int T, int Mkv) {
  __shared__ alignas(32) bf16_t sK[32 * 64];       // [key][dh]
  __shared__ alignas(32) bf16_t sVt[64 * 32];      // [dh][key]
  __shared__ alignas(32) bf16_t sP[4 * 16 * 32];   // per-wave P patch

  const int tid = threadIdx.x;
  const int wave = tid >> 5;
  const int lane = tid & 31;
  const int h = blockIdx.y;
  const int b = blockIdx.z;
  const int q0 = blockIdx.x * 64 + wave * 16;

  const bf16_t* Qb = Q + ((size_t)b * T + q0) * 768 + h * 64;
  v16bf aq0 = load_frag_a(Qb, 768);       // dh 0..31
  v16bf aq1 = load_frag_a(Qb + 32, 768);  // dh 32..63

  v8f o0 = {}, o1 = {}, o2 = {}, o3 = {};
  float mrow[8], lrow[8];
#pragma unroll
  for (int i = 0; i < 8; ++i) { mrow[i] = -1e30f; lrow[i] = 0.0f; }

  const int srow = tid >> 2;        // 0..31  (key row for staging)
  const int sseg = (tid & 3) << 4;  // 0/16/32/48 (dh segment)
  const int rb = (lane >> 4) << 3;
  const int cl = lane & 15;
  bf16_t* pw = sP + wave * (16 * 32);

  for (int kc = 0; kc < Mkv; kc += 32) {
    const bf16_t* Kc = Km + ((size_t)b * Mkv + kc) * 768 + h * 64;
    const bf16_t* Vc = Vm + ((size_t)b * Mkv + kc) * 768 + h * 64;
    *(v16bf*)(sK + srow * 64 + sseg) = *(const v16bf*)(Kc + (size_t)srow * 768 + sseg);
    v16bf tv = *(const v16bf*)(Vc + (size_t)srow * 768 + sseg);
#pragma unroll
    for (int j = 0; j < 16; ++j) sVt[(sseg + j) * 32 + srow] = tv[j];
    __syncthreads();

    // S = Q K^T : 16 queries x 32 keys (two 16-col tiles, K over dh=64)
    v8f s0 = {}, s1 = {};
    s0 = wmma_bf16(aq0, load_frag_bt(sK + 0, 64), s0);
    s0 = wmma_bf16(aq1, load_frag_bt(sK + 32, 64), s0);
    s1 = wmma_bf16(aq0, load_frag_bt(sK + 16 * 64, 64), s1);
    s1 = wmma_bf16(aq1, load_frag_bt(sK + 16 * 64 + 32, 64), s1);

    // online softmax; row r=rb+i is spread over the 16-lane half-group
#pragma unroll
    for (int i = 0; i < 8; ++i) {
      float v0 = s0[i] * 0.125f;  // 1/sqrt(64)
      float v1 = s1[i] * 0.125f;
      float t = fmaxf(v0, v1);
      t = fmaxf(t, __shfl_xor(t, 1, 32));
      t = fmaxf(t, __shfl_xor(t, 2, 32));
      t = fmaxf(t, __shfl_xor(t, 4, 32));
      t = fmaxf(t, __shfl_xor(t, 8, 32));
      float mn = fmaxf(mrow[i], t);
      float sc = __expf(mrow[i] - mn);
      float p0 = __expf(v0 - mn);
      float p1 = __expf(v1 - mn);
      float rs = p0 + p1;
      rs += __shfl_xor(rs, 1, 32);
      rs += __shfl_xor(rs, 2, 32);
      rs += __shfl_xor(rs, 4, 32);
      rs += __shfl_xor(rs, 8, 32);
      lrow[i] = lrow[i] * sc + rs;
      mrow[i] = mn;
      o0[i] *= sc; o1[i] *= sc; o2[i] *= sc; o3[i] *= sc;
      pw[(rb + i) * 32 + cl] = (bf16_t)p0;        // C-layout -> LDS patch
      pw[(rb + i) * 32 + 16 + cl] = (bf16_t)p1;
    }

    // O += P * V  (A-frag of P from LDS patch; V as B-frag over 4 dh tiles)
    v16bf ap = load_frag_a(pw, 32);
    o0 = wmma_bf16(ap, load_frag_bt(sVt + 0 * 16 * 32, 32), o0);
    o1 = wmma_bf16(ap, load_frag_bt(sVt + 1 * 16 * 32, 32), o1);
    o2 = wmma_bf16(ap, load_frag_bt(sVt + 2 * 16 * 32, 32), o2);
    o3 = wmma_bf16(ap, load_frag_bt(sVt + 3 * 16 * 32, 32), o3);
    __syncthreads();
  }

  float g = gates ? gates[b * 12 + h] : 1.0f;  // post-softmax per-head gate
  bf16_t* Zb = Z + ((size_t)b * T + q0) * 768 + h * 64;
#pragma unroll
  for (int i = 0; i < 8; ++i) {
    float inv = g / lrow[i];
    Zb[(size_t)(rb + i) * 768 + 0 * 16 + cl] = (bf16_t)(o0[i] * inv);
    Zb[(size_t)(rb + i) * 768 + 1 * 16 + cl] = (bf16_t)(o1[i] * inv);
    Zb[(size_t)(rb + i) * 768 + 2 * 16 + cl] = (bf16_t)(o2[i] * inv);
    Zb[(size_t)(rb + i) * 768 + 3 * 16 + cl] = (bf16_t)(o3[i] * inv);
  }
}

// ---------------------------------------------------------------------------
// Small helper kernels
// ---------------------------------------------------------------------------
__global__ void ln_bf16_kernel(const float* __restrict__ x, const float* __restrict__ g,
                               const float* __restrict__ b, bf16_t* __restrict__ out,
                               int D) {
  __shared__ float s1[256], s2[256];
  size_t row = blockIdx.x;
  const float* xr = x + row * D;
  float a0 = 0.f, a1 = 0.f;
  for (int i = threadIdx.x; i < D; i += 256) { float v = xr[i]; a0 += v; a1 += v * v; }
  s1[threadIdx.x] = a0; s2[threadIdx.x] = a1;
  __syncthreads();
  for (int off = 128; off; off >>= 1) {
    if ((int)threadIdx.x < off) {
      s1[threadIdx.x] += s1[threadIdx.x + off];
      s2[threadIdx.x] += s2[threadIdx.x + off];
    }
    __syncthreads();
  }
  float mu = s1[0] / D;
  float var = s2[0] / D - mu * mu;
  float rs = rsqrtf(var + 1e-5f);
  for (int i = threadIdx.x; i < D; i += 256)
    out[row * D + i] = (bf16_t)((xr[i] - mu) * rs * g[i] + b[i]);
}

__global__ void f32_to_bf16_kernel(const float* __restrict__ s, bf16_t* __restrict__ d,
                                   size_t n) {
  size_t i = (size_t)blockIdx.x * blockDim.x + threadIdx.x;
  size_t st = (size_t)gridDim.x * blockDim.x;
  for (; i < n; i += st) d[i] = (bf16_t)s[i];
}

__global__ void f32_to_bf16_sel_kernel(const float* __restrict__ base,
                                       const int* __restrict__ idx, size_t per,
                                       bf16_t* __restrict__ d, size_t n) {
  const float* s = base + (size_t)idx[0] * per;
  size_t i = (size_t)blockIdx.x * blockDim.x + threadIdx.x;
  size_t st = (size_t)gridDim.x * blockDim.x;
  for (; i < n; i += st) d[i] = (bf16_t)s[i];
}

__global__ void zero_bf16_kernel(bf16_t* __restrict__ d, size_t n) {
  size_t i = (size_t)blockIdx.x * blockDim.x + threadIdx.x;
  size_t st = (size_t)gridDim.x * blockDim.x;
  for (; i < n; i += st) d[i] = (bf16_t)0.0f;
}

__global__ void gate_kernel(const float* __restrict__ motif, const float* __restrict__ gw,
                            const float* __restrict__ gb, float* __restrict__ gates) {
  int t = threadIdx.x;
  if (t < 24) {
    int b = t / 12, h = t % 12;
    float s = gb[h];
    for (int i = 0; i < 8; ++i) s += motif[b * 8 + i] * gw[i * 12 + h];
    gates[b * 12 + h] = 1.0f / (1.0f + expf(-s));
  }
}

// ---------------------------------------------------------------------------
// Launch
// ---------------------------------------------------------------------------
extern "C" void kernel_launch(void* const* d_in, const int* in_sizes, int n_in,
                              void* d_out, int out_size, void* d_ws, size_t ws_size,
                              hipStream_t stream) {
  (void)in_sizes; (void)n_in; (void)out_size; (void)ws_size;
  const int B = 2, T = 2048, MKV = 2048, D = 768, H = 12, DFF = 3072;
  const int BT = B * T, BM = B * MKV;

  const float* x     = (const float*)d_in[0];
  const float* mem   = (const float*)d_in[1];
  const float* motif = (const float*)d_in[2];
  const int*   aidx  = (const int*)d_in[3];
  const float* ln1g = (const float*)d_in[4];  const float* ln1b = (const float*)d_in[5];
  const float* ln2g = (const float*)d_in[6];  const float* ln2b = (const float*)d_in[7];
  const float* ln3g = (const float*)d_in[8];  const float* ln3b = (const float*)d_in[9];
  const float* c_wq = (const float*)d_in[10]; const float* c_bq = (const float*)d_in[11];
  const float* c_wk = (const float*)d_in[12]; const float* c_bk = (const float*)d_in[13];
  const float* c_wv = (const float*)d_in[14]; const float* c_bv = (const float*)d_in[15];
  const float* c_wo = (const float*)d_in[16]; const float* c_bo = (const float*)d_in[17];
  const float* s_wq = (const float*)d_in[18]; const float* s_bq = (const float*)d_in[19];
  const float* s_wk = (const float*)d_in[20]; const float* s_bk = (const float*)d_in[21];
  const float* s_wv = (const float*)d_in[22]; const float* s_bv = (const float*)d_in[23];
  const float* s_wo = (const float*)d_in[24]; const float* s_bo = (const float*)d_in[25];
  const float* gw   = (const float*)d_in[26]; const float* gb   = (const float*)d_in[27];
  const float* fc1w = (const float*)d_in[28]; const float* fc1b = (const float*)d_in[29];
  const float* fc2w = (const float*)d_in[30]; const float* fc2b = (const float*)d_in[31];
  const float* adA  = (const float*)d_in[32]; const float* adB  = (const float*)d_in[33];
  float* out = (float*)d_out;

  // workspace arena (256B-aligned slices)
  size_t off = 0;
  auto alloc = [&](size_t elems, size_t esz) -> void* {
    void* p = (char*)d_ws + off;
    off += (elems * esz + 255) & ~(size_t)255;
    return p;
  };
  bf16_t* w_cq = (bf16_t*)alloc((size_t)D * D, 2);
  bf16_t* w_ck = (bf16_t*)alloc((size_t)D * D, 2);
  bf16_t* w_cv = (bf16_t*)alloc((size_t)D * D, 2);
  bf16_t* w_co = (bf16_t*)alloc((size_t)D * D, 2);
  bf16_t* w_sq = (bf16_t*)alloc((size_t)D * D, 2);
  bf16_t* w_sk = (bf16_t*)alloc((size_t)D * D, 2);
  bf16_t* w_sv = (bf16_t*)alloc((size_t)D * D, 2);
  bf16_t* w_so = (bf16_t*)alloc((size_t)D * D, 2);
  bf16_t* w_f1 = (bf16_t*)alloc((size_t)D * DFF, 2);
  bf16_t* w_f2 = (bf16_t*)alloc((size_t)DFF * D, 2);
  bf16_t* w_aA = (bf16_t*)alloc((size_t)D * 16, 2);
  bf16_t* w_aB = (bf16_t*)alloc((size_t)32 * D, 2);  // K padded 16 -> 32
  bf16_t* hbuf = (bf16_t*)alloc((size_t)BT * D, 2);
  bf16_t* memb = (bf16_t*)alloc((size_t)BM * D, 2);
  bf16_t* qb   = (bf16_t*)alloc((size_t)BT * D, 2);
  bf16_t* kb   = (bf16_t*)alloc((size_t)BM * D, 2);
  bf16_t* vb   = (bf16_t*)alloc((size_t)BM * D, 2);
  bf16_t* zb   = (bf16_t*)alloc((size_t)BT * D, 2);
  bf16_t* hh   = (bf16_t*)alloc((size_t)BT * DFF, 2);
  bf16_t* tl   = (bf16_t*)alloc((size_t)BT * 32, 2);  // LoRA mid, padded cols
  float*  xres = (float*)alloc((size_t)BT * D, 4);
  float*  gts  = (float*)alloc((size_t)B * H, 4);

  auto conv = [&](const float* s, bf16_t* d, size_t n) {
    int blocks = (int)((n + 255) / 256); if (blocks > 2048) blocks = 2048;
    f32_to_bf16_kernel<<<blocks, 256, 0, stream>>>(s, d, n);
  };
  auto gemm = [&](const bf16_t* A, const bf16_t* W, const float* bias,
                  const float* resid, float* oF, bf16_t* oB, int M_, int N_, int K_,
                  int ldc_, bool gelu) {
    dim3 grid((N_ + 63) / 64, (M_ + 127) / 128);
    if (gelu)
      gemm_bf16_kernel<true><<<grid, 256, 0, stream>>>(A, W, bias, resid, oF, oB, M_, N_, K_, ldc_);
    else
      gemm_bf16_kernel<false><<<grid, 256, 0, stream>>>(A, W, bias, resid, oF, oB, M_, N_, K_, ldc_);
  };

  // weights -> bf16
  conv(c_wq, w_cq, (size_t)D * D); conv(c_wk, w_ck, (size_t)D * D);
  conv(c_wv, w_cv, (size_t)D * D); conv(c_wo, w_co, (size_t)D * D);
  conv(s_wq, w_sq, (size_t)D * D); conv(s_wk, w_sk, (size_t)D * D);
  conv(s_wv, w_sv, (size_t)D * D); conv(s_wo, w_so, (size_t)D * D);
  conv(fc1w, w_f1, (size_t)D * DFF); conv(fc2w, w_f2, (size_t)DFF * D);
  zero_bf16_kernel<<<96, 256, 0, stream>>>(w_aB, (size_t)32 * D);
  f32_to_bf16_sel_kernel<<<48, 256, 0, stream>>>(adB, aidx, (size_t)16 * D, w_aB, (size_t)16 * D);
  f32_to_bf16_sel_kernel<<<48, 256, 0, stream>>>(adA, aidx, (size_t)D * 16, w_aA, (size_t)D * 16);
  zero_bf16_kernel<<<512, 256, 0, stream>>>(tl, (size_t)BT * 32);
  gate_kernel<<<1, 32, 0, stream>>>(motif, gw, gb, gts);
  conv(mem, memb, (size_t)BM * D);

  // ---- block 1: cross attention ----
  ln_bf16_kernel<<<BT, 256, 0, stream>>>(x, ln1g, ln1b, hbuf, D);
  gemm(hbuf, w_cq, c_bq, nullptr, nullptr, qb, BT, D, D, D, false);
  gemm(memb, w_ck, c_bk, nullptr, nullptr, kb, BM, D, D, D, false);
  gemm(memb, w_cv, c_bv, nullptr, nullptr, vb, BM, D, D, D, false);
  flash_attn_kernel<<<dim3(T / 64, H, B), 128, 0, stream>>>(qb, kb, vb, nullptr, zb, T, MKV);
  gemm(zb, w_co, c_bo, x, xres, nullptr, BT, D, D, D, false);  // x1 = x + attn

  // ---- block 2: motif-gated self attention ----
  ln_bf16_kernel<<<BT, 256, 0, stream>>>(xres, ln2g, ln2b, hbuf, D);
  gemm(hbuf, w_sq, s_bq, nullptr, nullptr, qb, BT, D, D, D, false);
  gemm(hbuf, w_sk, s_bk, nullptr, nullptr, kb, BT, D, D, D, false);
  gemm(hbuf, w_sv, s_bv, nullptr, nullptr, vb, BT, D, D, D, false);
  flash_attn_kernel<<<dim3(T / 64, H, B), 128, 0, stream>>>(qb, kb, vb, gts, zb, T, T);
  gemm(zb, w_so, s_bo, xres, xres, nullptr, BT, D, D, D, false);  // x2 = x1 + attn

  // ---- block 3: LoRA FFN ----
  ln_bf16_kernel<<<BT, 256, 0, stream>>>(xres, ln3g, ln3b, hbuf, D);
  gemm(hbuf, w_f1, fc1b, nullptr, nullptr, hh, BT, DFF, D, DFF, true);   // GELU
  gemm(hbuf, w_aA, nullptr, nullptr, nullptr, tl, BT, 16, D, 32, false); // h3@adA
  gemm(hh, w_f2, fc2b, xres, out, nullptr, BT, D, DFF, D, false);        // x2+fc2
  gemm(tl, w_aB, nullptr, out, out, nullptr, BT, D, 32, D, false);       // +lora
}
